// AdaptiveLRLinearWithChannel_36481452212287
// MI455X (gfx1250) — matrix operand order repaired and checked by reading, not compiled, not run on hardware
//
#include <hip/hip_runtime.h>

#define IN_SZ   256
#define OUT_SZ  256
#define RANK    60
#define NSEL    256
#define BATCH   1024
#define WTOT    (IN_SZ * OUT_SZ)   // 65536 per-channel weight elements

typedef _Float16 H;
typedef __attribute__((ext_vector_type(16))) _Float16 v16h;
typedef __attribute__((ext_vector_type(8)))  float    v8f;

// ---------------------------------------------------------------------------
// Stage 1:  Wsel[n, :] = U[idx[n], :] @ V          (f32 inputs -> f16 output)
// GEMM  M = NSEL(256) x N = 65536, K = 64 (60 zero-padded)
// block = 128 threads (4 waves); wave computes a 16(M) x 64(N) strip.
// All loads are unconditional (indices clamped in-bounds); padding is done
// with branchless value-selects so no EXEC-masked scalar fallbacks appear.
// ---------------------------------------------------------------------------
__global__ __launch_bounds__(128)
void build_weights_kernel(const float* __restrict__ U,
                          const float* __restrict__ V,
                          const int*   __restrict__ idx,
                          H* __restrict__ Wsel)
{
    const int lane = threadIdx.x & 31;
    const int wave = threadIdx.x >> 5;
    const int half = lane >> 4;     // which K-half this lane supplies for A
    const int l16  = lane & 15;

    const int mrow = blockIdx.y * 16 + l16;          // selected-channel row
    const int chan = idx[mrow];
    const float* Urow = U + (size_t)chan * RANK;

    const int colw = blockIdx.x * 256 + wave * 64;   // wave's first N column

    v8f acc[4] = {v8f{}, v8f{}, v8f{}, v8f{}};

    #pragma unroll
    for (int ks = 0; ks < 2; ++ks) {
        const int kbase = ks * 32;

        // ---- A tile 16x32 of U_sel: element j -> K = kbase + (j<8 ?
        //      half*8+j : 16 + half*8 + j-8).  Load clamped, select after.
        v16h a;
        #pragma unroll
        for (int j = 0; j < 16; ++j) {
            const int k  = kbase + ((j < 8) ? (half * 8 + j)
                                            : (16 + half * 8 + (j - 8)));
            const int kc = (k < RANK) ? k : (RANK - 1);   // always in-bounds
            const float uv = Urow[kc];                    // unconditional load
            a[j] = (H)((k < RANK) ? uv : 0.0f);           // branchless pad
        }

        // ---- B tiles 32x16: lane holds column l16, element j -> same K as A.
        //      Row clamped to a valid V row; padded rows zeroed by select.
        const float* Vp  = V + colw + l16;
        #pragma unroll
        for (int nt = 0; nt < 4; ++nt) {
            v16h b;
            #pragma unroll
            for (int j = 0; j < 16; ++j) {
                const int k  = kbase + ((j < 8) ? (half * 8 + j)
                                                : (16 + half * 8 + (j - 8)));
                const int kc = (k < RANK) ? k : (RANK - 1);
                const float vv = Vp[(size_t)kc * WTOT + nt * 16]; // unconditional load
                b[j] = (H)((k < RANK) ? vv : 0.0f);       // branchless pad
            }
            acc[nt] = __builtin_amdgcn_wmma_f32_16x16x32_f16(
                          false, a, false, b, (short)0, acc[nt], false, false);
        }
    }

    // D store: VGPR j, lane -> (row = j + 8*half, col = l16) within tile
    const int row0 = blockIdx.y * 16;
    #pragma unroll
    for (int nt = 0; nt < 4; ++nt) {
        const int col = colw + nt * 16 + l16;
        #pragma unroll
        for (int j = 0; j < 8; ++j) {
            const int row = row0 + j + 8 * half;
            Wsel[(size_t)row * WTOT + col] = (H)acc[nt][j];
        }
    }
}

// ---------------------------------------------------------------------------
// Stage 2:  out[n] = x[n] @ Wsel[n] + bias[idx[n]]
// Per n: GEMM M = 1024, N = 256, K = 256.
// block = 128 threads (4 waves); WG computes a 64x64 tile of out[n];
// wave computes a 16(M) x 64(N) strip (4 accumulator tiles), K unrolled.
// ---------------------------------------------------------------------------
__global__ __launch_bounds__(128)
void gemm_kernel(const float* __restrict__ x,
                 const H*     __restrict__ Wsel,
                 const float* __restrict__ bias,
                 const int*   __restrict__ idx,
                 float*       __restrict__ out)
{
    const int lane = threadIdx.x & 31;
    const int wave = threadIdx.x >> 5;
    const int half = lane >> 4;
    const int l16  = lane & 15;

    const int n     = blockIdx.z;                          // selected channel
    const int mbase = blockIdx.y * 64 + wave * 16;         // wave's M strip
    const int colb  = blockIdx.x * 64;                     // WG column base

    const float* xrow = x + ((size_t)n * BATCH + (mbase + l16)) * IN_SZ;
    const H*     Wn   = Wsel + (size_t)n * WTOT;

    v8f acc[4] = {v8f{}, v8f{}, v8f{}, v8f{}};

    #pragma unroll
    for (int kbase = 0; kbase < IN_SZ; kbase += 32) {
        // A tile 16x32 from x (f32 -> f16): two contiguous 8-element runs
        v16h a;
        const int k0 = kbase + half * 8;
        #pragma unroll
        for (int j = 0; j < 8; ++j) a[j]     = (H)xrow[k0 + j];
        #pragma unroll
        for (int j = 0; j < 8; ++j) a[8 + j] = (H)xrow[k0 + 16 + j];

        // B tiles 32x16 from Wsel (f16): lane holds column l16, K as in A
        const H* Bp = Wn + colb + l16;
        #pragma unroll
        for (int nt = 0; nt < 4; ++nt) {
            v16h b;
            #pragma unroll
            for (int j = 0; j < 8; ++j) b[j]     = Bp[(size_t)(k0 + j) * OUT_SZ + nt * 16];
            #pragma unroll
            for (int j = 0; j < 8; ++j) b[8 + j] = Bp[(size_t)(k0 + 16 + j) * OUT_SZ + nt * 16];
            acc[nt] = __builtin_amdgcn_wmma_f32_16x16x32_f16(
                          false, a, false, b, (short)0, acc[nt], false, false);
        }
    }

    // Epilogue: add bias (depends only on output column), store f32
    const int    chan = idx[n];
    const float* brow = bias + (size_t)chan * OUT_SZ;
    float* obase = out + (size_t)n * BATCH * OUT_SZ;
    #pragma unroll
    for (int nt = 0; nt < 4; ++nt) {
        const int   col = colb + nt * 16 + l16;
        const float bv  = brow[col];
        #pragma unroll
        for (int j = 0; j < 8; ++j) {
            const int row = mbase + j + 8 * half;
            obase[(size_t)row * OUT_SZ + col] = acc[nt][j] + bv;
        }
    }
}

// ---------------------------------------------------------------------------
extern "C" void kernel_launch(void* const* d_in, const int* in_sizes, int n_in,
                              void* d_out, int out_size, void* d_ws, size_t ws_size,
                              hipStream_t stream)
{
    (void)in_sizes; (void)n_in; (void)out_size; (void)ws_size;

    const float* x    = (const float*)d_in[0];
    const int*   idx  = (const int*)  d_in[1];
    const float* U    = (const float*)d_in[2];
    const float* V    = (const float*)d_in[3];
    const float* bias = (const float*)d_in[4];
    float*       out  = (float*)d_out;
    H*           Wsel = (H*)d_ws;      // NSEL * WTOT * 2B = 32 MB scratch

    // Stage 1: synthesize selected per-channel weights once (f16 in ws)
    build_weights_kernel<<<dim3(WTOT / 256, NSEL / 16), 128, 0, stream>>>(
        U, V, idx, Wsel);

    // Stage 2: 256 batched GEMMs + bias
    gemm_kernel<<<dim3(OUT_SZ / 64, BATCH / 64, NSEL), 128, 0, stream>>>(
        x, Wsel, bias, idx, out);
}
